// CrossAttention_65008624992354
// MI455X (gfx1250) — compile-verified, matrix operand order checked by reading, not run on hardware
//
#include <hip/hip_runtime.h>
#include <hip/hip_bf16.h>

typedef _Float16 v16h __attribute__((ext_vector_type(16)));
typedef _Float16 v8h  __attribute__((ext_vector_type(8)));
typedef float    v8f  __attribute__((ext_vector_type(8)));

#define B_  2
#define N_  4096
#define M_  4096
#define QD_ 512
#define CD_ 768
#define H_  8
#define D_  64
#define IN_ 512

static __device__ __forceinline__ v8f vzero() {
  v8f r;
#pragma unroll
  for (int i = 0; i < 8; ++i) r[i] = 0.0f;
  return r;
}

static __device__ __forceinline__ v8f wmma16(v16h a, v16h b, v8f c) {
  return __builtin_amdgcn_wmma_f32_16x16x32_f16(false, a, false, b, (short)0, c,
                                                false, false);
}

static __device__ __forceinline__ v16h cat8(v8h lo, v8h hi) {
  v16h r;
#pragma unroll
  for (int i = 0; i < 8; ++i) { r[i] = lo[i]; r[8 + i] = hi[i]; }
  return r;
}

// 16-bit A/B fragment (16x32 tile): this lane's row pointer; two 16B chunks at
// K = half*8..half*8+7 and K = 16+half*8..16+half*8+7.
static __device__ __forceinline__ v16h ldfrag_h(const _Float16* rowp, int half) {
  v8h lo = *(const v8h*)(rowp + half * 8);
  v8h hi = *(const v8h*)(rowp + 16 + half * 8);
  return cat8(lo, hi);
}

static __device__ __forceinline__ v16h ldfrag_f32(const float* rowp, int half) {
  const float* p0 = rowp + half * 8;
  const float* p1 = rowp + 16 + half * 8;
  v16h r;
#pragma unroll
  for (int i = 0; i < 8; ++i) { r[i] = (_Float16)p0[i]; r[8 + i] = (_Float16)p1[i]; }
  return r;
}

static __device__ __forceinline__ float redmax16(float v) {
#pragma unroll
  for (int off = 1; off < 16; off <<= 1) v = fmaxf(v, __shfl_xor(v, off, 32));
  return v;
}
static __device__ __forceinline__ float redsum16(float v) {
#pragma unroll
  for (int off = 1; off < 16; off <<= 1) v += __shfl_xor(v, off, 32);
  return v;
}

// ---------------- weight transpose + f32 -> f16 ----------------
__global__ void prep_weights(const float* __restrict__ Wq, const float* __restrict__ Wk,
                             const float* __restrict__ Wv, const float* __restrict__ Wo,
                             _Float16* __restrict__ WqT, _Float16* __restrict__ WkT,
                             _Float16* __restrict__ WvT, _Float16* __restrict__ WoT) {
  int i = blockIdx.x * blockDim.x + threadIdx.x;
  if (i < IN_ * QD_) {                 // WqT[j][k] = Wq[k][j]
    int j = i / QD_, k = i - j * QD_;
    WqT[i] = (_Float16)Wq[(size_t)k * IN_ + j];
  }
  if (i < IN_ * CD_) {                 // WkT/WvT[j][k] = W[k][j]
    int j = i / CD_, k = i - j * CD_;
    WkT[i] = (_Float16)Wk[(size_t)k * IN_ + j];
    WvT[i] = (_Float16)Wv[(size_t)k * IN_ + j];
  }
  if (i < QD_ * IN_) {                 // WoT[j][k] = Wo[k][j]
    int j = i / IN_, k = i - j * IN_;
    WoT[i] = (_Float16)Wo[(size_t)k * QD_ + j];
  }
}

// ---------------- generic projection GEMM: C(f16) = A(f32) * Bt(f16)^T ----------------
__global__ __launch_bounds__(256) void proj_kernel(const float* __restrict__ A,
                                                   const _Float16* __restrict__ Bt,
                                                   _Float16* __restrict__ C,
                                                   int rows, int cols, int kdim) {
  const int lane = threadIdx.x & 31;
  const int wid = blockIdx.x * 8 + (threadIdx.x >> 5);
  const int ctiles = cols >> 4;
  const int mt = wid / ctiles;
  const int nt = wid % ctiles;
  if (mt * 16 >= rows) return;
  const int r16 = lane & 15, half = lane >> 4;
  const float* Arow = A + (size_t)(mt * 16 + r16) * kdim;
  const _Float16* Brow = Bt + (size_t)(nt * 16 + r16) * kdim;
  v8f acc = vzero();
  for (int k = 0; k < kdim; k += 32)
    acc = wmma16(ldfrag_f32(Arow + k, half), ldfrag_h(Brow + k, half), acc);
#pragma unroll
  for (int r = 0; r < 8; ++r) {
    int row = mt * 16 + ((lane < 16) ? r : 8 + r);
    C[(size_t)row * cols + nt * 16 + r16] = (_Float16)acc[r];
  }
}

// ---------------- V projection with transposed store: Vt[b][h][d][m] ----------------
__global__ __launch_bounds__(256) void projv_kernel(const float* __restrict__ A,
                                                    const _Float16* __restrict__ Bt,
                                                    _Float16* __restrict__ Vt) {
  const int lane = threadIdx.x & 31;
  const int wid = blockIdx.x * 8 + (threadIdx.x >> 5);
  const int ctiles = IN_ >> 4;
  const int mt = wid / ctiles;
  const int nt = wid % ctiles;
  if (mt * 16 >= B_ * M_) return;
  const int r16 = lane & 15, half = lane >> 4;
  const float* Arow = A + (size_t)(mt * 16 + r16) * CD_;
  const _Float16* Brow = Bt + (size_t)(nt * 16 + r16) * CD_;
  v8f acc = vzero();
  for (int k = 0; k < CD_; k += 32)
    acc = wmma16(ldfrag_f32(Arow + k, half), ldfrag_h(Brow + k, half), acc);
  const int b = (mt * 16) / M_;
  const int m0 = mt * 16 - b * M_ + ((lane < 16) ? 0 : 8);  // comps are m0..m0+7
  const int j = nt * 16 + r16;
  const int h = j >> 6, d = j & 63;
  v8h pk;
#pragma unroll
  for (int r = 0; r < 8; ++r) pk[r] = (_Float16)acc[r];
  *(v8h*)(Vt + ((size_t)(b * H_ + h) * D_ + d) * M_ + m0) = pk;
}

// ---------------- fused flash cross-attention (LDS-staged K/V) ----------------
__global__ __launch_bounds__(256) void attn_kernel(const _Float16* __restrict__ Qh,
                                                   const _Float16* __restrict__ Kh,
                                                   const _Float16* __restrict__ Vt,
                                                   const int* __restrict__ mask,
                                                   _Float16* __restrict__ Oh) {
  __shared__ _Float16 Kt[32 * 64];       // chunk of K:  [m_local][d]
  __shared__ _Float16 Vtile[64 * 32];    // chunk of Vt: [d][m_local]
  __shared__ _Float16 Plds[8][16 * 32];  // per-wave P staging (C-layout -> A-fragment)

  const int tid = threadIdx.x;
  const int lane = tid & 31;
  const int wave = tid >> 5;
  const int b = blockIdx.x >> 3;
  const int h = blockIdx.x & 7;
  const int ntile = blockIdx.y * 128 + wave * 16;   // 16 query rows per wave
  const int r16 = lane & 15;
  const int half = lane >> 4;

  const _Float16* Qrow = Qh + ((size_t)(b * N_ + ntile + r16)) * IN_ + h * D_;
  const v16h aq0 = ldfrag_h(Qrow, half);        // d = 0..31
  const v16h aq1 = ldfrag_h(Qrow + 32, half);   // d = 32..63

  const _Float16* Kbase = Kh + (size_t)b * M_ * IN_ + h * D_;
  const _Float16* Vbase = Vt + ((size_t)(b * H_ + h)) * D_ * M_;
  const int* mk = mask + b * M_;
  _Float16* pw = Plds[wave];

  // cooperative stage mapping: 16B per thread per tile
  const int krow = tid >> 3, kseg = (tid & 7) * 8;   // K chunk  [32][64]
  const int vrow = tid >> 2, vseg = (tid & 3) * 8;   // Vt chunk [64][32]
  const _Float16* kg = Kbase + (size_t)krow * IN_ + kseg;
  const _Float16* vg = Vbase + (size_t)vrow * M_ + vseg;

  float mrun[8], lrun[8];
  v8f acc[4];
#pragma unroll
  for (int r = 0; r < 8; ++r) { mrun[r] = -3.0e38f; lrun[r] = 0.0f; }
#pragma unroll
  for (int t = 0; t < 4; ++t) acc[t] = vzero();

  // prefetch chunk 0 into registers
  v8h kreg = *(const v8h*)kg;
  v8h vreg = *(const v8h*)vg;

  for (int c = 0; c < M_; c += 32) {
    __syncthreads();                         // all LDS reads of prev chunk done
    *(v8h*)&Kt[krow * 64 + kseg] = kreg;
    *(v8h*)&Vtile[vrow * 32 + vseg] = vreg;
    __syncthreads();                         // staged data visible
    if (c + 32 < M_) {                       // prefetch next chunk (overlaps compute)
      kreg = *(const v8h*)(kg + (size_t)(c + 32) * IN_);
      vreg = *(const v8h*)(vg + (c + 32));
    }

    // ---- S = Q K^T for 16x32 tile (two 16x16 subtiles, K-depth 64) ----
    v8f s0 = vzero(), s1 = vzero();
    {
      const _Float16* Krow0 = Kt + r16 * 64;
      const _Float16* Krow1 = Kt + (16 + r16) * 64;
      s0 = wmma16(aq0, ldfrag_h(Krow0, half), s0);
      s0 = wmma16(aq1, ldfrag_h(Krow0 + 32, half), s0);
      s1 = wmma16(aq0, ldfrag_h(Krow1, half), s1);
      s1 = wmma16(aq1, ldfrag_h(Krow1 + 32, half), s1);
    }
    const bool mv0 = mk[c + r16] != 0;
    const bool mv1 = mk[c + 16 + r16] != 0;

    float cmax[8];
#pragma unroll
    for (int r = 0; r < 8; ++r) {
      float a0 = mv0 ? s0[r] * 0.125f : -3.0e38f;  // scale = 1/sqrt(64)
      float a1 = mv1 ? s1[r] * 0.125f : -3.0e38f;
      s0[r] = a0; s1[r] = a1;
      cmax[r] = fmaxf(a0, a1);
    }
#pragma unroll
    for (int r = 0; r < 8; ++r) cmax[r] = redmax16(cmax[r]);

    float alpha[8];
#pragma unroll
    for (int r = 0; r < 8; ++r) {
      float nm = fmaxf(mrun[r], cmax[r]);
      alpha[r] = __expf(mrun[r] - nm);
      mrun[r] = nm;
    }

    float csum[8];
#pragma unroll
    for (int r = 0; r < 8; ++r) {
      float p0 = mv0 ? __expf(s0[r] - mrun[r]) : 0.0f;  // masked -> exactly 0
      float p1 = mv1 ? __expf(s1[r] - mrun[r]) : 0.0f;
      csum[r] = p0 + p1;
      const int row = (lane < 16) ? r : 8 + r;
      pw[row * 32 + r16] = (_Float16)p0;
      pw[row * 32 + 16 + r16] = (_Float16)p1;
    }
#pragma unroll
    for (int r = 0; r < 8; ++r) {
      csum[r] = redsum16(csum[r]);
      lrun[r] = lrun[r] * alpha[r] + csum[r];
    }
#pragma unroll
    for (int t = 0; t < 4; ++t)
#pragma unroll
      for (int r = 0; r < 8; ++r) acc[t][r] *= alpha[r];

    __builtin_amdgcn_wave_barrier();   // keep wave-private LDS writes before reads
    const v16h pa = ldfrag_h(pw + r16 * 32, half);

    // ---- O += P V : 4 d-tiles, K-depth 32 (this chunk) ----
#pragma unroll
    for (int t = 0; t < 4; ++t) {
      const _Float16* Vrow = Vtile + (t * 16 + r16) * 32;
      acc[t] = wmma16(pa, ldfrag_h(Vrow, half), acc[t]);
    }
  }

  float inv[8];
#pragma unroll
  for (int r = 0; r < 8; ++r) inv[r] = (lrun[r] > 0.0f) ? 1.0f / lrun[r] : 0.0f;
#pragma unroll
  for (int t = 0; t < 4; ++t) {
#pragma unroll
    for (int r = 0; r < 8; ++r) {
      const int row = ntile + ((lane < 16) ? r : 8 + r);
      const int col = h * D_ + t * 16 + r16;
      Oh[(size_t)(b * N_ + row) * IN_ + col] = (_Float16)(acc[t][r] * inv[r]);
    }
  }
}

// ---------------- output projection: out(f32) = Ah(f16) * WoT^T + bo ----------------
__global__ __launch_bounds__(256) void outproj_kernel(const _Float16* __restrict__ Ah,
                                                      const _Float16* __restrict__ WoT,
                                                      const float* __restrict__ bo,
                                                      float* __restrict__ Out) {
  const int lane = threadIdx.x & 31;
  const int wid = blockIdx.x * 8 + (threadIdx.x >> 5);
  const int ctiles = QD_ >> 4;
  const int mt = wid / ctiles;
  const int nt = wid % ctiles;
  if (mt * 16 >= B_ * N_) return;
  const int r16 = lane & 15, half = lane >> 4;
  const _Float16* Arow = Ah + (size_t)(mt * 16 + r16) * IN_;
  const _Float16* Brow = WoT + (size_t)(nt * 16 + r16) * IN_;
  v8f acc = vzero();
  for (int k = 0; k < IN_; k += 32)
    acc = wmma16(ldfrag_h(Arow + k, half), ldfrag_h(Brow + k, half), acc);
  const int col = nt * 16 + r16;
  const float bias = bo[col];
#pragma unroll
  for (int r = 0; r < 8; ++r) {
    int row = mt * 16 + ((lane < 16) ? r : 8 + r);
    Out[(size_t)row * QD_ + col] = acc[r] + bias;
  }
}

extern "C" void kernel_launch(void* const* d_in, const int* in_sizes, int n_in,
                              void* d_out, int out_size, void* d_ws, size_t ws_size,
                              hipStream_t stream) {
  (void)in_sizes; (void)n_in; (void)out_size; (void)ws_size;
  const float* x   = (const float*)d_in[0];
  const float* ctx = (const float*)d_in[1];
  const int*   msk = (const int*)d_in[2];
  const float* Wq  = (const float*)d_in[3];
  const float* Wk  = (const float*)d_in[4];
  const float* Wv  = (const float*)d_in[5];
  const float* Wo  = (const float*)d_in[6];
  const float* bo  = (const float*)d_in[7];
  float* out = (float*)d_out;

  char* w = (char*)d_ws;
  _Float16* WqT = (_Float16*)w; w += (size_t)IN_ * QD_ * 2;
  _Float16* WkT = (_Float16*)w; w += (size_t)IN_ * CD_ * 2;
  _Float16* WvT = (_Float16*)w; w += (size_t)IN_ * CD_ * 2;
  _Float16* WoT = (_Float16*)w; w += (size_t)QD_ * IN_ * 2;
  _Float16* Qh  = (_Float16*)w; w += (size_t)B_ * N_ * IN_ * 2;
  _Float16* Kh  = (_Float16*)w; w += (size_t)B_ * M_ * IN_ * 2;
  _Float16* Vt  = (_Float16*)w; w += (size_t)B_ * H_ * D_ * M_ * 2;
  _Float16* Ah  = (_Float16*)w;

  prep_weights<<<(IN_ * CD_ + 255) / 256, 256, 0, stream>>>(Wq, Wk, Wv, Wo,
                                                            WqT, WkT, WvT, WoT);
  const int gemm_blocks = ((B_ * N_ / 16) * (IN_ / 16)) / 8;  // 2048 blocks, 8 waves each
  proj_kernel<<<gemm_blocks, 256, 0, stream>>>(x,   WqT, Qh, B_ * N_, IN_, QD_);
  proj_kernel<<<gemm_blocks, 256, 0, stream>>>(ctx, WkT, Kh, B_ * M_, IN_, CD_);
  projv_kernel<<<gemm_blocks, 256, 0, stream>>>(ctx, WvT, Vt);
  attn_kernel<<<dim3(B_ * H_, N_ / 128), 256, 0, stream>>>(Qh, Kh, Vt, msk, Ah);
  outproj_kernel<<<gemm_blocks, 256, 0, stream>>>(Ah, WoT, bo, out);
}
